// SwinDiTBlock_45758581572279
// MI455X (gfx1250) — compile-verified, matrix operand order checked by reading, not run on hardware
//
#include <hip/hip_runtime.h>
#include <cmath>

typedef __attribute__((ext_vector_type(16))) __bf16 v16bf;
typedef __attribute__((ext_vector_type(8)))  __bf16 v8bf;
typedef __attribute__((ext_vector_type(8)))  float  v8f;

#define TOK   65536   // 16 * 64 * 64 tokens
#define DIMC  512
#define HEADS 16
#define HD    32
#define MLP   2048
#define SCALEQ 0.17677669529663687f   // 32^-0.5

// ---------- helpers ----------
__device__ __forceinline__ __bf16 f2bf(float f){
  unsigned u = __builtin_bit_cast(unsigned, f);
  unsigned r = (u + 0x7FFFu + ((u >> 16) & 1u)) >> 16;
  unsigned short s = (unsigned short)r;
  return __builtin_bit_cast(__bf16, s);
}

__device__ __forceinline__ v8f wmma_bf16(v16bf a, v16bf b, v8f c){
  return __builtin_amdgcn_wmma_f32_16x16x32_bf16(false, a, false, b, (short)0, c,
                                                 false, false);
}

// Load a 16x32 bf16 A-fragment (or B-fragment, B stored as [N,K] row-major)
// per ISA 7.12.2: lanes 0-15 hold rows 0-15 with K={0..7,16..23},
// lanes 16-31 hold the same rows with K={8..15,24..31}.
// Two 16-byte chunks -> global_load_b128 / ds_load_b128.
__device__ __forceinline__ v16bf load_frag(const __bf16* base, long row0, int ldk,
                                           int k0, int lane){
  const __bf16* p = base + (row0 + (lane & 15)) * (long)ldk + k0 + ((lane >> 4) << 3);
  v8bf lo = *(const v8bf*)(p);
  v8bf hi = *(const v8bf*)(p + 16);
  return __builtin_shufflevector(lo, hi, 0, 1, 2, 3, 4, 5, 6, 7,
                                         8, 9, 10, 11, 12, 13, 14, 15);
}

// ---------- weight conversion ----------
__global__ void cvt_bf16_kernel(const float* __restrict__ src, __bf16* __restrict__ dst, long n){
  long i = (long)blockIdx.x * blockDim.x + threadIdx.x;
  long stride = (long)gridDim.x * blockDim.x;
  for (; i < n; i += stride) dst[i] = f2bf(src[i]);
}

// ---------- adaLN: ada = silu(t_emb) @ ada_w^T + ada_b, [16, 2048] ----------
__global__ __launch_bounds__(128) void ada_kernel(const float* __restrict__ t_emb,
                                                  const float* __restrict__ W,
                                                  const float* __restrict__ bsrc,
                                                  float* __restrict__ out){
  __shared__ float s[512];
  int b = blockIdx.x;
  for (int i = threadIdx.x; i < 512; i += 128){
    float v = t_emb[b * 512 + i];
    s[i] = v / (1.f + __expf(-v));
  }
  __syncthreads();
  int j = blockIdx.y * 128 + threadIdx.x;
  const float* wr = W + (long)j * 512;
  float acc = bsrc[j];
  for (int k = 0; k < 512; ++k) acc += s[k] * wr[k];
  out[b * 2048 + j] = acc;
}

// ---------- layernorm + modulate (+ optional roll(-4,-4) + window partition) ----------
template<bool WINDOWED>
__global__ __launch_bounds__(256) void norm_mod_kernel(const float* __restrict__ x,
                                                       const float* __restrict__ g,
                                                       const float* __restrict__ be,
                                                       const float* __restrict__ ada,
                                                       int shift_off, int scale_off,
                                                       __bf16* __restrict__ out){
  __shared__ float s1[256], s2[256];
  int t = blockIdx.x, tid = threadIdx.x;
  int b = t >> 12;
  const float* src;
  if constexpr (WINDOWED){
    int nw = (t >> 6) & 63, i = t & 63;
    int hp = (nw >> 3) * 8 + (i >> 3), wp = (nw & 7) * 8 + (i & 7);
    int h = (hp + 4) & 63, w = (wp + 4) & 63;       // roll(-4,-4) gather
    src = x + (((long)b * 64 + h) * 64 + w) * DIMC;
  } else {
    src = x + (long)t * DIMC;
  }
  float v0 = src[tid], v1 = src[tid + 256];
  s1[tid] = v0 + v1;
  s2[tid] = v0 * v0 + v1 * v1;
  __syncthreads();
  for (int off = 128; off > 0; off >>= 1){
    if (tid < off){ s1[tid] += s1[tid + off]; s2[tid] += s2[tid + off]; }
    __syncthreads();
  }
  float mean = s1[0] * (1.f / 512.f);
  float var  = s2[0] * (1.f / 512.f) - mean * mean;
  float rs   = rsqrtf(var + 1e-5f);
  const float* arow = ada + b * 2048;
  #pragma unroll
  for (int e = 0; e < 2; ++e){
    int c = tid + e * 256;
    float v = e ? v1 : v0;
    float y = (v - mean) * rs * g[c] + be[c];
    y = y * (1.f + arow[scale_off + c]) + arow[shift_off + c];
    out[(long)t * DIMC + c] = f2bf(y);
  }
}

// ---------- generic WMMA GEMM: C[M,N] = A[M,K](bf16) @ W[N,K]^T + bias ----------
// Block tile 128x128, 8 waves, each wave a 32x64 tile (2x4 WMMA frags).
// MODE 0: QKV scatter (q scaled, v transposed)   MODE 1: proj + unwindow + residual
// MODE 2: fc1 + GELU -> bf16                      MODE 3: fc2 + residual -> f32
template<int MODE>
__global__ __launch_bounds__(256) void wmma_gemm(const __bf16* __restrict__ A,
                                                 const __bf16* __restrict__ W,
                                                 const float* __restrict__ bias,
                                                 int K,
                                                 __bf16* __restrict__ oQ,
                                                 __bf16* __restrict__ oK,
                                                 __bf16* __restrict__ oVT,
                                                 const float* __restrict__ resid,
                                                 float* __restrict__ of32,
                                                 __bf16* __restrict__ obf){
  int lane = threadIdx.x & 31, wid = threadIdx.x >> 5;
  int wm = wid >> 1, wn = wid & 1;          // 4 x 2 wave grid
  long row0 = (long)blockIdx.x * 128 + wm * 32;
  int  col0 = blockIdx.y * 128 + wn * 64;
  const v8f vz = {0.f,0.f,0.f,0.f,0.f,0.f,0.f,0.f};
  v8f acc[2][4] = {{vz, vz, vz, vz}, {vz, vz, vz, vz}};
  for (int k0 = 0; k0 < K; k0 += 32){
    v16bf a0 = load_frag(A, row0,      K, k0, lane);
    v16bf a1 = load_frag(A, row0 + 16, K, k0, lane);
    v16bf b0 = load_frag(W, col0,      K, k0, lane);
    v16bf b1 = load_frag(W, col0 + 16, K, k0, lane);
    v16bf b2 = load_frag(W, col0 + 32, K, k0, lane);
    v16bf b3 = load_frag(W, col0 + 48, K, k0, lane);
    acc[0][0] = wmma_bf16(a0, b0, acc[0][0]);
    acc[0][1] = wmma_bf16(a0, b1, acc[0][1]);
    acc[0][2] = wmma_bf16(a0, b2, acc[0][2]);
    acc[0][3] = wmma_bf16(a0, b3, acc[0][3]);
    acc[1][0] = wmma_bf16(a1, b0, acc[1][0]);
    acc[1][1] = wmma_bf16(a1, b1, acc[1][1]);
    acc[1][2] = wmma_bf16(a1, b2, acc[1][2]);
    acc[1][3] = wmma_bf16(a1, b3, acc[1][3]);
  }
  #pragma unroll
  for (int mi = 0; mi < 2; ++mi){
    #pragma unroll
    for (int ni = 0; ni < 4; ++ni){
      int colb = col0 + ni * 16 + (lane & 15);
      float bval = bias[colb];
      #pragma unroll
      for (int r = 0; r < 8; ++r){
        int row = (int)row0 + mi * 16 + r + ((lane >> 4) << 3);
        float v = acc[mi][ni][r] + bval;
        if constexpr (MODE == 0){
          int p = colb >> 9, hd = (colb >> 5) & 15, d = colb & 31;
          int b_ = row >> 6, rr = row & 63;
          long base = (long)b_ * 16 + hd;
          if (p == 0)      oQ [(base * 64 + rr) * 32 + d] = f2bf(v * SCALEQ);
          else if (p == 1) oK [(base * 64 + rr) * 32 + d] = f2bf(v);
          else             oVT[(base * 32 + d)  * 64 + rr] = f2bf(v);
        } else if constexpr (MODE == 1){
          int b = row >> 12, nw = (row >> 6) & 63, i = row & 63;
          int hp = (nw >> 3) * 8 + (i >> 3), wp = (nw & 7) * 8 + (i & 7);
          int h = (hp + 4) & 63, w = (wp + 4) & 63;   // roll(+4,+4) scatter
          long idx = (((long)b * 64 + h) * 64 + w) * DIMC + colb;
          of32[idx] = resid[idx] + v;
        } else if constexpr (MODE == 2){
          float gl = 0.5f * v * (1.f + erff(v * 0.70710678118654752f));
          obf[(long)row * MLP + colb] = f2bf(gl);
        } else {
          long idx = (long)row * DIMC + colb;
          of32[idx] = resid[idx] + v;
        }
      }
    }
  }
}

// ---------- windowed attention, one block per (window, head), 4 waves ----------
__device__ __forceinline__ int region(int h){ return h < 56 ? 0 : (h < 60 ? 1 : 2); }

__global__ __launch_bounds__(128) void attn_kernel(const __bf16* __restrict__ Q,
                                                   const __bf16* __restrict__ Km,
                                                   const __bf16* __restrict__ VT,
                                                   const float* __restrict__ rpb,
                                                   __bf16* __restrict__ out){
  __shared__ float  sc[64][68];
  __shared__ __bf16 pr[64][64];
  __shared__ float  rpbs[225];      // this head's relative-position bias table
  int bh   = blockIdx.x;
  int head = bh & 15;
  int b_   = bh >> 4;
  int nw   = b_ & 63;
  int wr = nw >> 3, wc = nw & 7;
  int lane = threadIdx.x & 31, wid = threadIdx.x >> 5;
  int tid  = threadIdx.x;
  const __bf16* q  = Q  + (long)bh * 64 * 32;
  const __bf16* k  = Km + (long)bh * 64 * 32;
  const __bf16* vt = VT + (long)bh * 32 * 64;
  int m0 = wid * 16;

  // stage bias column for this head into LDS (225 floats)
  for (int i = tid; i < 225; i += 128) rpbs[i] = rpb[i * HEADS + head];
  __syncthreads();

  // scores = (q*scale) @ k^T  -- one WMMA per 16x16 tile (K = HD = 32)
  v16bf aq = load_frag(q, m0, 32, 0, lane);
  for (int nt = 0; nt < 4; ++nt){
    v16bf bk = load_frag(k, nt * 16, 32, 0, lane);
    v8f c = {0.f,0.f,0.f,0.f,0.f,0.f,0.f,0.f};
    c = wmma_bf16(aq, bk, c);
    int col = nt * 16 + (lane & 15);
    #pragma unroll
    for (int r = 0; r < 8; ++r){
      int row = m0 + r + ((lane >> 4) << 3);
      float v = c[r];
      // relative position bias (index computed inline, table in LDS)
      int dr = (row >> 3) - (col >> 3), dc = (row & 7) - (col & 7);
      v += rpbs[(dr + 7) * 15 + (dc + 7)];
      // shifted-window mask (region labels computed inline)
      int hi = wr * 8 + (row >> 3), wi = wc * 8 + (row & 7);
      int hj = wr * 8 + (col >> 3), wj = wc * 8 + (col & 7);
      int li = region(hi) * 3 + region(wi);
      int lj = region(hj) * 3 + region(wj);
      if (li != lj) v -= 100.f;
      sc[row][col] = v;
    }
  }
  __syncthreads();

  // per-row softmax -> bf16 probs
  if (tid < 64){
    float mx = -1e30f;
    for (int j = 0; j < 64; ++j) mx = fmaxf(mx, sc[tid][j]);
    float sum = 0.f;
    float e[64];
    for (int j = 0; j < 64; ++j){ e[j] = __expf(sc[tid][j] - mx); sum += e[j]; }
    float inv = 1.f / sum;
    for (int j = 0; j < 64; ++j) pr[tid][j] = f2bf(e[j] * inv);
  }
  __syncthreads();

  // out = probs @ V  (V stored transposed so B-frag is K-contiguous)
  const v8f vz = {0.f,0.f,0.f,0.f,0.f,0.f,0.f,0.f};
  v8f acc[2] = {vz, vz};
  for (int kt = 0; kt < 2; ++kt){
    v16bf ap = load_frag(&pr[0][0], m0, 64, kt * 32, lane);
    #pragma unroll
    for (int nt = 0; nt < 2; ++nt){
      v16bf bv = load_frag(vt, nt * 16, 64, kt * 32, lane);
      acc[nt] = wmma_bf16(ap, bv, acc[nt]);
    }
  }
  #pragma unroll
  for (int nt = 0; nt < 2; ++nt){
    int col = nt * 16 + (lane & 15);
    #pragma unroll
    for (int r = 0; r < 8; ++r){
      int row = m0 + r + ((lane >> 4) << 3);
      out[((long)(b_ * 64 + row)) * DIMC + head * 32 + col] = f2bf(acc[nt][r]);
    }
  }
}

// ---------- host launcher ----------
extern "C" void kernel_launch(void* const* d_in, const int* in_sizes, int n_in,
                              void* d_out, int out_size, void* d_ws, size_t ws_size,
                              hipStream_t stream) {
  const float* x       = (const float*)d_in[0];
  const float* t_emb   = (const float*)d_in[1];
  const float* norm1_g = (const float*)d_in[2];
  const float* norm1_b = (const float*)d_in[3];
  const float* qkv_w   = (const float*)d_in[4];
  const float* qkv_b   = (const float*)d_in[5];
  const float* proj_w  = (const float*)d_in[6];
  const float* proj_b  = (const float*)d_in[7];
  const float* rpb     = (const float*)d_in[8];
  const float* norm2_g = (const float*)d_in[9];
  const float* norm2_b = (const float*)d_in[10];
  const float* fc1_w   = (const float*)d_in[11];
  const float* fc1_b   = (const float*)d_in[12];
  const float* fc2_w   = (const float*)d_in[13];
  const float* fc2_b   = (const float*)d_in[14];
  const float* ada_w   = (const float*)d_in[15];
  const float* ada_b   = (const float*)d_in[16];
  float* out = (float*)d_out;

  char* p = (char*)d_ws;
  auto alloc = [&](size_t bytes) -> char* {
    char* r = p; p += (bytes + 255) & ~(size_t)255; return r;
  };
  __bf16* wqkv  = (__bf16*)alloc((size_t)1536 * 512 * 2);
  __bf16* wproj = (__bf16*)alloc((size_t)512 * 512 * 2);
  __bf16* wfc1  = (__bf16*)alloc((size_t)2048 * 512 * 2);
  __bf16* wfc2  = (__bf16*)alloc((size_t)512 * 2048 * 2);
  float*  ada   = (float*)alloc((size_t)16 * 2048 * 4);
  // big region R1 (256 MiB): xw | q | k | vT during attention, then reused as fc1 hidden
  char* R1 = alloc((size_t)TOK * MLP * 2);
  __bf16* xw  = (__bf16*)R1;
  __bf16* qb  = (__bf16*)(R1 + (size_t)TOK * DIMC * 2);
  __bf16* kb  = (__bf16*)(R1 + (size_t)TOK * DIMC * 4);
  __bf16* vTb = (__bf16*)(R1 + (size_t)TOK * DIMC * 6);
  __bf16* hid = (__bf16*)R1;
  __bf16* attn_out = (__bf16*)alloc((size_t)TOK * DIMC * 2); // later reused as act2
  __bf16* act2 = attn_out;
  float* x1 = (float*)alloc((size_t)TOK * DIMC * 4);

  // 1) weights -> bf16
  cvt_bf16_kernel<<<1024, 256, 0, stream>>>(qkv_w, wqkv, (long)1536 * 512);
  cvt_bf16_kernel<<<512,  256, 0, stream>>>(proj_w, wproj, (long)512 * 512);
  cvt_bf16_kernel<<<1024, 256, 0, stream>>>(fc1_w, wfc1, (long)2048 * 512);
  cvt_bf16_kernel<<<1024, 256, 0, stream>>>(fc2_w, wfc2, (long)512 * 2048);
  // 2) adaLN params
  ada_kernel<<<dim3(16, 16), 128, 0, stream>>>(t_emb, ada_w, ada_b, ada);
  // 3) norm1 + modulate + roll + window-partition -> xw (bf16)
  norm_mod_kernel<true><<<TOK, 256, 0, stream>>>(x, norm1_g, norm1_b, ada, 0, 512, xw);
  // 4) QKV GEMM (65536 x 1536 x 512), scatter q/k/vT
  wmma_gemm<0><<<dim3(TOK / 128, 1536 / 128), 256, 0, stream>>>(
      xw, wqkv, qkv_b, 512, qb, kb, vTb, nullptr, nullptr, nullptr);
  // 5) attention per (window, head)
  attn_kernel<<<TOK / 64 * HEADS, 128, 0, stream>>>(qb, kb, vTb, rpb, attn_out);
  // 6) proj GEMM + window-reverse + roll-back + residual -> x1 (f32)
  wmma_gemm<1><<<dim3(TOK / 128, 512 / 128), 256, 0, stream>>>(
      attn_out, wproj, proj_b, 512, nullptr, nullptr, nullptr, x, x1, nullptr);
  // 7) norm2 + modulate -> act2 (bf16, reuses attn_out buffer)
  norm_mod_kernel<false><<<TOK, 256, 0, stream>>>(x1, norm2_g, norm2_b, ada, 1024, 1536, act2);
  // 8) fc1 + GELU -> hid (bf16, reuses R1)
  wmma_gemm<2><<<dim3(TOK / 128, MLP / 128), 256, 0, stream>>>(
      act2, wfc1, fc1_b, 512, nullptr, nullptr, nullptr, nullptr, nullptr, hid);
  // 9) fc2 + residual -> d_out (f32)
  wmma_gemm<3><<<dim3(TOK / 128, 512 / 128), 256, 0, stream>>>(
      hid, wfc2, fc2_b, MLP, nullptr, nullptr, nullptr, x1, out, nullptr);
}
